// Attention_335007449334
// MI455X (gfx1250) — matrix operand order repaired and checked by reading, not compiled, not run on hardware
//
#include <hip/hip_runtime.h>
#include <hip/hip_bf16.h>
#include <math.h>

typedef __attribute__((ext_vector_type(16))) __bf16 bf16x16;
typedef __attribute__((ext_vector_type(8)))  __bf16 bf16x8;
typedef __attribute__((ext_vector_type(8)))  float  f32x8;

#define B_   4
#define S_   512
#define D_   4096
#define H_   32
#define HK_  8
#define HD_  128
#define AL_  64

// ---------------- WMMA helpers (CDNA5 wave32 layouts per ISA 7.12.2) -------

__device__ __forceinline__ f32x8 wmma_bf16(bf16x16 a, bf16x16 b, f32x8 c) {
  return __builtin_amdgcn_wmma_f32_16x16x32_bf16(false, a, false, b, (short)0,
                                                 c, false, false);
}

// A fragment: 16x32 (MxK), row-major source, leading dim lda (elements).
// lane<16: row=lane, K=0..7 & 16..23 ; lane>=16: row=lane-16, K=8..15 & 24..31
__device__ __forceinline__ bf16x16 load_a_frag(const __bf16* p, int lda) {
  const int l = threadIdx.x & 31;
  const __bf16* r0 = p + (size_t)(l & 15) * lda + ((l >> 4) << 3);
  bf16x8 lo = *(const bf16x8*)r0;
  bf16x8 hi = *(const bf16x8*)(r0 + 16);
  bf16x16 a;
#pragma unroll
  for (int i = 0; i < 8; ++i) { a[i] = lo[i]; a[8 + i] = hi[i]; }
  return a;
}

// B fragment: 32x16 (KxN), where storage is "column n of B is a contiguous
// row of length K" (i.e. B^T row-major, leading dim ldn). lane<16: col=lane,
// K=0..7 & 16..23 ; lane>=16: col=lane-16, K=8..15 & 24..31.
__device__ __forceinline__ bf16x16 load_b_fragT(const __bf16* p, int ldn) {
  const int l = threadIdx.x & 31;
  const __bf16* r0 = p + (size_t)(l & 15) * ldn + ((l >> 4) << 3);
  bf16x8 lo = *(const bf16x8*)r0;
  bf16x8 hi = *(const bf16x8*)(r0 + 16);
  bf16x16 b;
#pragma unroll
  for (int i = 0; i < 8; ++i) { b[i] = lo[i]; b[8 + i] = hi[i]; }
  return b;
}

// ---------------- precision-conversion kernels -----------------------------

__global__ void cast_f32_bf16(const float* __restrict__ in,
                              __bf16* __restrict__ out, int n) {
  int i = blockIdx.x * blockDim.x + threadIdx.x;
  if (i < n) out[i] = (__bf16)in[i];
}

// in: (R x C) f32 row-major  ->  out: (C x R) bf16 row-major
__global__ void transpose_cast(const float* __restrict__ in,
                               __bf16* __restrict__ out, int R, int C) {
  __shared__ float tile[32][33];
  int c = blockIdx.x * 32 + threadIdx.x;
  int r = blockIdx.y * 32 + threadIdx.y;
#pragma unroll
  for (int j = 0; j < 32; j += 8)
    tile[threadIdx.y + j][threadIdx.x] = in[(size_t)(r + j) * C + c];
  __syncthreads();
  int ro = blockIdx.x * 32 + threadIdx.y;
  int co = blockIdx.y * 32 + threadIdx.x;
#pragma unroll
  for (int j = 0; j < 32; j += 8)
    out[(size_t)(ro + j) * R + co] = (__bf16)tile[threadIdx.x][threadIdx.y + j];
}

// ---------------- fused GEMM (bf16 WMMA, f32 accum) ------------------------
// Block tile 64x128, 4 waves in 2x2; per-wave tile 32x64 (2x4 WMMA tiles).
// Per k-step: 12 b128 loads feed 8 WMMAs (1.5 loads/WMMA vs 2.0 for 32x32).
// epilogue modes
#define MODE_Q   0   // rope + qscale, layout (b,H,s,hd)
#define MODE_K   1   // rope,          layout (b,HK,s,hd)
#define MODE_VT  2   //                layout (b,HK,hd,s)  (transposed)
#define MODE_AK  3   //                layout (b,HK,s,hd)
#define MODE_AVT 4   //                layout (b,HK,hd,s)  (transposed)
#define MODE_F32 5   // plain f32 row-major (final output)

__global__ void __launch_bounds__(128) gemm_bf16_wmma(
    const __bf16* __restrict__ A, const __bf16* __restrict__ BT,
    void* __restrict__ Out, int M, int N, int K, int mode,
    const float* __restrict__ cosT, const float* __restrict__ sinT,
    int Srows, int Hout, float qscale)
{
  const int wave = threadIdx.x >> 5;
  const int l    = threadIdx.x & 31;
  const int row0 = blockIdx.y * 64  + (wave >> 1) * 32;
  const int col0 = blockIdx.x * 128 + (wave & 1) * 64;

  f32x8 acc[2][4] = {};
  for (int k0 = 0; k0 < K; k0 += 32) {
    bf16x16 a0 = load_a_frag(A + (size_t)row0 * K + k0, K);
    bf16x16 a1 = load_a_frag(A + (size_t)(row0 + 16) * K + k0, K);
#pragma unroll
    for (int ni = 0; ni < 4; ++ni) {
      bf16x16 b = load_b_fragT(BT + (size_t)(col0 + ni * 16) * K + k0, K);
      acc[0][ni] = wmma_bf16(a0, b, acc[0][ni]);
      acc[1][ni] = wmma_bf16(a1, b, acc[1][ni]);
    }
  }

  const int rsub = (l >> 4) << 3;
  const int csub = l & 15;
#pragma unroll
  for (int mi = 0; mi < 2; ++mi)
#pragma unroll
    for (int ni = 0; ni < 4; ++ni)
#pragma unroll
      for (int r = 0; r < 8; ++r) {
        int row = row0 + mi * 16 + r + rsub;
        int col = col0 + ni * 16 + csub;
        float v = acc[mi][ni][r];
        if (mode <= MODE_K) {  // RoPE: pair partner lives in lane^1 (wave32)
          float vp = __shfl_xor(v, 1, 32);
          int spos = row & (Srows - 1);
          int hd   = col & (HD_ - 1);
          float cs = cosT[spos * (HD_ / 2) + (hd >> 1)];
          float sn = sinT[spos * (HD_ / 2) + (hd >> 1)];
          v = (col & 1) ? fmaf(vp, sn, v * cs) : fmaf(-vp, sn, v * cs);
          if (mode == MODE_Q) v *= qscale;  // fold 1/sqrt(HD) into Q
        }
        if (mode == MODE_F32) {
          ((float*)Out)[(size_t)row * N + col] = v;
        } else {
          int b = row / Srows, s = row - b * Srows;
          int h = col >> 7, hd = col & (HD_ - 1);
          size_t idx;
          if (mode == MODE_VT || mode == MODE_AVT)
            idx = ((size_t)(b * Hout + h) * HD_ + hd) * Srows + s;
          else
            idx = ((size_t)(b * Hout + h) * Srows + s) * HD_ + hd;
          ((__bf16*)Out)[idx] = (__bf16)v;
        }
      }
}

// ---------------- flash-style causal attention + adapter -------------------
// grid: (S/64 q-tiles, B*H). block = 128 threads = 4 waves; wave owns 16 rows.

__global__ void __launch_bounds__(128) attn_wmma(
    const __bf16* __restrict__ Q,  const __bf16* __restrict__ Km,
    const __bf16* __restrict__ Vt, const __bf16* __restrict__ AK,
    const __bf16* __restrict__ AVt, const float* __restrict__ gate,
    __bf16* __restrict__ Outp)
{
  const int bh = blockIdx.y;
  const int b  = bh >> 5;
  const int h  = bh & 31;
  const int hk = h >> 2;                 // GQA 4:1
  const int qbase = blockIdx.x * 64;
  const int wave  = threadIdx.x >> 5;
  const int l     = threadIdx.x & 31;
  const int qrow0 = qbase + wave * 16;
  const int rsub  = (l >> 4) << 3;       // C-frag row offset of this lane-half
  const int csub  = l & 15;              // C-frag column of this lane

  const __bf16* q   = Q   + ((size_t)(b * H_  + h ) * S_ ) * HD_;
  const __bf16* k   = Km  + ((size_t)(b * HK_ + hk) * S_ ) * HD_;
  const __bf16* vt  = Vt  + ((size_t)(b * HK_ + hk) * HD_) * S_;
  const __bf16* ak  = AK  + ((size_t)(b * HK_ + hk) * AL_) * HD_;
  const __bf16* avt = AVt + ((size_t)(b * HK_ + hk) * HD_) * AL_;

  __shared__ __bf16 pbuf[4][16 * 64];    // per-wave prob staging (C->A relayout)
  __bf16* myp = pbuf[wave];

  bf16x16 qf[4];                         // 16 q rows x HD=128 (4 K-steps)
#pragma unroll
  for (int i = 0; i < 4; ++i)
    qf[i] = load_a_frag(q + (size_t)qrow0 * HD_ + i * 32, HD_);

  f32x8 o[8] = {};                       // 16 x 128 f32 accumulators
  float m[8], lsum[8];
#pragma unroll
  for (int r = 0; r < 8; ++r) { m[r] = -1e30f; lsum[r] = 0.f; }

  for (int k0 = 0; k0 <= qbase; k0 += 64) {
    f32x8 s[4] = {};                     // 16 rows x 64 keys
#pragma unroll
    for (int t = 0; t < 4; ++t)
#pragma unroll
      for (int kk = 0; kk < 4; ++kk)
        s[t] = wmma_bf16(qf[kk],
                 load_b_fragT(k + (size_t)(k0 + t * 16) * HD_ + kk * 32, HD_),
                 s[t]);

    if (k0 == qbase) {                   // causal mask on diagonal tile only
#pragma unroll
      for (int t = 0; t < 4; ++t)
#pragma unroll
        for (int r = 0; r < 8; ++r) {
          int qr  = qrow0 + r + rsub;
          int key = k0 + t * 16 + csub;
          if (key > qr) s[t][r] = -1e30f;
        }
    }

    // online softmax: row stats live in 16-lane halves matching C-frag rows
#pragma unroll
    for (int r = 0; r < 8; ++r) {
      float x = fmaxf(fmaxf(s[0][r], s[1][r]), fmaxf(s[2][r], s[3][r]));
#pragma unroll
      for (int mk = 8; mk >= 1; mk >>= 1) x = fmaxf(x, __shfl_xor(x, mk, 32));
      float mnew  = fmaxf(m[r], x);
      float alpha = __expf(m[r] - mnew);
      float psum  = 0.f;
#pragma unroll
      for (int t = 0; t < 4; ++t) {
        float p = __expf(s[t][r] - mnew); s[t][r] = p; psum += p;
      }
#pragma unroll
      for (int mk = 8; mk >= 1; mk >>= 1) psum += __shfl_xor(psum, mk, 32);
      lsum[r] = lsum[r] * alpha + psum;
      m[r]    = mnew;
#pragma unroll
      for (int n = 0; n < 8; ++n) o[n][r] *= alpha;
    }

    // stage probs 16x64 row-major bf16 in wave-private LDS, reload as A-frags
#pragma unroll
    for (int t = 0; t < 4; ++t)
#pragma unroll
      for (int r = 0; r < 8; ++r)
        myp[(r + rsub) * 64 + t * 16 + csub] = (__bf16)s[t][r];
    asm volatile("s_wait_dscnt 0" ::: "memory");

#pragma unroll
    for (int t2 = 0; t2 < 2; ++t2) {
      bf16x16 pf = load_a_frag(myp + t2 * 32, 64);
#pragma unroll
      for (int n = 0; n < 8; ++n)
        o[n] = wmma_bf16(pf,
                 load_b_fragT(vt + (size_t)(n * 16) * S_ + k0 + t2 * 32, S_),
                 o[n]);
    }
    asm volatile("s_wait_dscnt 0" ::: "memory");
  }

  // normalize main attention
#pragma unroll
  for (int r = 0; r < 8; ++r) {
    float inv = 1.f / lsum[r];
#pragma unroll
    for (int n = 0; n < 8; ++n) o[n][r] *= inv;
  }

  // adapter branch: exact softmax over 64 keys, scaled by tanh(gate[h])
  {
    f32x8 sa[4] = {};
#pragma unroll
    for (int t = 0; t < 4; ++t)
#pragma unroll
      for (int kk = 0; kk < 4; ++kk)
        sa[t] = wmma_bf16(qf[kk],
                  load_b_fragT(ak + (size_t)(t * 16) * HD_ + kk * 32, HD_),
                  sa[t]);

    float tg = tanhf(gate[h]);
#pragma unroll
    for (int r = 0; r < 8; ++r) {
      float x = fmaxf(fmaxf(sa[0][r], sa[1][r]), fmaxf(sa[2][r], sa[3][r]));
#pragma unroll
      for (int mk = 8; mk >= 1; mk >>= 1) x = fmaxf(x, __shfl_xor(x, mk, 32));
      float psum = 0.f;
#pragma unroll
      for (int t = 0; t < 4; ++t) {
        float p = __expf(sa[t][r] - x); sa[t][r] = p; psum += p;
      }
#pragma unroll
      for (int mk = 8; mk >= 1; mk >>= 1) psum += __shfl_xor(psum, mk, 32);
      float sc = tg / psum;
#pragma unroll
      for (int t = 0; t < 4; ++t) sa[t][r] *= sc;
    }

#pragma unroll
    for (int t = 0; t < 4; ++t)
#pragma unroll
      for (int r = 0; r < 8; ++r)
        myp[(r + rsub) * 64 + t * 16 + csub] = (__bf16)sa[t][r];
    asm volatile("s_wait_dscnt 0" ::: "memory");

#pragma unroll
    for (int t2 = 0; t2 < 2; ++t2) {
      bf16x16 pf = load_a_frag(myp + t2 * 32, 64);
#pragma unroll
      for (int n = 0; n < 8; ++n)
        o[n] = wmma_bf16(pf,
                 load_b_fragT(avt + (size_t)(n * 16) * AL_ + t2 * 32, AL_),
                 o[n]);
    }
  }

  // write attn_out as (B, S, H*HD) bf16 for the final projection GEMM
#pragma unroll
  for (int n = 0; n < 8; ++n)
#pragma unroll
    for (int r = 0; r < 8; ++r) {
      int srow = qrow0 + r + rsub;
      int hd   = n * 16 + csub;
      Outp[(size_t)(b * S_ + srow) * (H_ * HD_) + h * HD_ + hd] =
          (__bf16)o[n][r];
    }
}

// ---------------- launcher -------------------------------------------------

extern "C" void kernel_launch(void* const* d_in, const int* in_sizes, int n_in,
                              void* d_out, int out_size, void* d_ws,
                              size_t ws_size, hipStream_t stream)
{
  const float* x    = (const float*)d_in[0];
  const float* adp  = (const float*)d_in[1];
  /* d_in[2] = mask: causal mask applied analytically in attn_wmma */
  const float* fcos = (const float*)d_in[3];
  const float* fsin = (const float*)d_in[4];
  const float* wq   = (const float*)d_in[5];
  const float* wk   = (const float*)d_in[6];
  const float* wv   = (const float*)d_in[7];
  const float* wo   = (const float*)d_in[8];
  const float* gate = (const float*)d_in[9];
  float* out = (float*)d_out;
  (void)in_sizes; (void)n_in; (void)out_size; (void)ws_size;

  char* ws = (char*)d_ws;
  size_t off = 0;
  auto alloc = [&](size_t elems) {
    void* p = ws + off;
    off += (elems * sizeof(__bf16) + 255) & ~(size_t)255;
    return (__bf16*)p;
  };
  const int MX = B_ * S_;    // 2048 token rows
  const int MA = B_ * AL_;   // 256 adapter rows
  __bf16* xbf  = alloc((size_t)MX * D_);
  __bf16* abf  = alloc((size_t)MA * D_);
  __bf16* wqT  = alloc((size_t)D_ * D_);
  __bf16* wkT  = alloc((size_t)(HK_ * HD_) * D_);
  __bf16* wvT  = alloc((size_t)(HK_ * HD_) * D_);
  __bf16* woT  = alloc((size_t)D_ * D_);
  __bf16* qh   = alloc((size_t)B_ * H_  * S_  * HD_);
  __bf16* kh   = alloc((size_t)B_ * HK_ * S_  * HD_);
  __bf16* vtt  = alloc((size_t)B_ * HK_ * HD_ * S_);
  __bf16* akh  = alloc((size_t)B_ * HK_ * AL_ * HD_);
  __bf16* avt  = alloc((size_t)B_ * HK_ * HD_ * AL_);
  __bf16* attn = alloc((size_t)MX * D_);

  const float qscale = 0.08838834764831845f;  // 1/sqrt(128)

  { int n = MX * D_;
    cast_f32_bf16<<<(n + 255) / 256, 256, 0, stream>>>(x, xbf, n); }
  { int n = MA * D_;
    cast_f32_bf16<<<(n + 255) / 256, 256, 0, stream>>>(adp, abf, n); }

  transpose_cast<<<dim3(D_ / 32, D_ / 32), dim3(32, 8), 0, stream>>>(wq, wqT, D_, D_);
  transpose_cast<<<dim3((HK_ * HD_) / 32, D_ / 32), dim3(32, 8), 0, stream>>>(wk, wkT, D_, HK_ * HD_);
  transpose_cast<<<dim3((HK_ * HD_) / 32, D_ / 32), dim3(32, 8), 0, stream>>>(wv, wvT, D_, HK_ * HD_);
  transpose_cast<<<dim3(D_ / 32, D_ / 32), dim3(32, 8), 0, stream>>>(wo, woT, D_, D_);

  gemm_bf16_wmma<<<dim3(D_ / 128, MX / 64), 128, 0, stream>>>(
      xbf, wqT, qh, MX, D_, D_, MODE_Q, fcos, fsin, S_, H_, qscale);
  gemm_bf16_wmma<<<dim3((HK_ * HD_) / 128, MX / 64), 128, 0, stream>>>(
      xbf, wkT, kh, MX, HK_ * HD_, D_, MODE_K, fcos, fsin, S_, HK_, 1.f);
  gemm_bf16_wmma<<<dim3((HK_ * HD_) / 128, MX / 64), 128, 0, stream>>>(
      xbf, wvT, vtt, MX, HK_ * HD_, D_, MODE_VT, nullptr, nullptr, S_, HK_, 1.f);
  gemm_bf16_wmma<<<dim3((HK_ * HD_) / 128, MA / 64), 128, 0, stream>>>(
      abf, wkT, akh, MA, HK_ * HD_, D_, MODE_AK, nullptr, nullptr, AL_, HK_, 1.f);
  gemm_bf16_wmma<<<dim3((HK_ * HD_) / 128, MA / 64), 128, 0, stream>>>(
      abf, wvT, avt, MA, HK_ * HD_, D_, MODE_AVT, nullptr, nullptr, AL_, HK_, 1.f);

  attn_wmma<<<dim3(S_ / 64, B_ * H_), 128, 0, stream>>>(
      qh, kh, vtt, akh, avt, gate, attn);

  gemm_bf16_wmma<<<dim3(D_ / 128, MX / 64), 128, 0, stream>>>(
      attn, woT, out, MX, D_, D_, MODE_F32, nullptr, nullptr, S_, H_, 1.f);
}
